// CorrelationVolumeWarpingQKV_6528350290335
// MI455X (gfx1250) — compile-verified, matrix operand order checked by reading, not run on hardware
//
#include <hip/hip_runtime.h>
#include <hip/hip_bf16.h>

// ---------------------------------------------------------------------------
// CorrelationVolumeWarpingQKV, fused flash-attention style for MI455X (gfx1250)
//
// Roofline: naive path materializes 512MB cvolume (~2GB traffic ~90us @23.3TB/s).
// Fused: ~70MB traffic -> WMMA-bound (~4.5M v_wmma_f32_16x16x32_f16).
// Scores are bounded (|q.k|<=1 after L2 norm) -> softmax without running max:
//   p = exp(s);  l = p . ones  (computed inside the PV WMMA via a ones-column);
//   max_score = max(p)/l  (one butterfly at kernel end).
// K/V tiles double-buffered in LDS, staged by the Tensor Data Mover when the
// builtin exists (pad feature provides the conflict-free LDS strides).
// ---------------------------------------------------------------------------

typedef __attribute__((ext_vector_type(16))) _Float16 v16h;
typedef __attribute__((ext_vector_type(8)))  _Float16 v8h;
typedef __attribute__((ext_vector_type(8)))  float    v8f;

#ifndef __has_builtin
#define __has_builtin(x) 0
#endif
#if __has_builtin(__builtin_amdgcn_tensor_load_to_lds) && \
    __has_builtin(__builtin_amdgcn_s_wait_tensorcnt)
#define USE_TDM 1
#else
#define USE_TDM 0
#endif

#define BATCH 8
#define DCH   128
#define NPIX  4096      // 64*64
#define OUTC  259       // 2*D + 3
#define VCH   132       // v1(128) + grid u,v(2) + ones(1) + pad(1)

// ======================= kernel 1: QKV prep ================================
#define PREP_TILE 32

__global__ __launch_bounds__(32) void qkv_prep(
    const float* __restrict__ x0, const float* __restrict__ x1,
    const float* __restrict__ Wq, const float* __restrict__ Wk,
    const float* __restrict__ Wv,
    _Float16* __restrict__ Qh, _Float16* __restrict__ Kh,
    _Float16* __restrict__ Vt, float* __restrict__ out)
{
    __shared__ float sx0[DCH][PREP_TILE];
    __shared__ float sx1[DCH][PREP_TILE];

    const int b  = blockIdx.y;
    const int n0 = blockIdx.x * PREP_TILE;
    const int t  = threadIdx.x;
    const int n  = n0 + t;

    for (int d = 0; d < DCH; ++d) {      // coalesced along n
        sx0[d][t] = x0[((size_t)b*DCH + d)*NPIX + n];
        sx1[d][t] = x1[((size_t)b*DCH + d)*NPIX + n];
    }
    __syncthreads();

    // pass 1: v0, v1 + sum of squares for q,k
    float sumq = 0.f, sumk = 0.f;
    for (int o = 0; o < DCH; ++o) {
        const float* wq = Wq + o*DCH;
        const float* wk = Wk + o*DCH;
        const float* wv = Wv + o*DCH;
        float aq = 0.f, ak = 0.f, a0 = 0.f, a1 = 0.f;
        for (int j = 0; j < DCH; ++j) {  // weights wave-uniform -> scalar loads
            float u0 = sx0[j][t], u1 = sx1[j][t];
            aq += wq[j]*u0; ak += wk[j]*u1;
            a0 += wv[j]*u0; a1 += wv[j]*u1;
        }
        aq += sx0[o][t]; ak += sx1[o][t];      // residuals
        a0 += sx0[o][t]; a1 += sx1[o][t];
        sumq += aq*aq; sumk += ak*ak;
        out[((size_t)b*OUTC + o)*NPIX + n] = a0;            // v0 fp32
        Vt [((size_t)b*VCH  + o)*NPIX + n] = (_Float16)a1;  // v1 channel-major
    }
    const float invq = 1.f / fmaxf(sqrtf(sumq), 1e-12f);
    const float invk = 1.f / fmaxf(sqrtf(sumk), 1e-12f);

    // pass 2: recompute q,k (cheap) and store normalized f16 row-major in d
    for (int o = 0; o < DCH; ++o) {
        const float* wq = Wq + o*DCH;
        const float* wk = Wk + o*DCH;
        float aq = 0.f, ak = 0.f;
        for (int j = 0; j < DCH; ++j) {
            aq += wq[j]*sx0[j][t];
            ak += wk[j]*sx1[j][t];
        }
        aq += sx0[o][t]; ak += sx1[o][t];
        Qh[((size_t)(b*NPIX + n))*DCH + o] = (_Float16)(aq*invq);
        Kh[((size_t)(b*NPIX + n))*DCH + o] = (_Float16)(ak*invk);
    }

    // augmented V channels: grid u,v ('ij' meshgrid), ones column (row-sum), pad
    const float uu = -1.f + (2.f/63.f)*(float)(n >> 6);
    const float vv = -1.f + (2.f/63.f)*(float)(n & 63);
    Vt[((size_t)b*VCH + 128)*NPIX + n] = (_Float16)uu;
    Vt[((size_t)b*VCH + 129)*NPIX + n] = (_Float16)vv;
    Vt[((size_t)b*VCH + 130)*NPIX + n] = (_Float16)1.0f;   // ones -> l = sum(p)
    Vt[((size_t)b*VCH + 131)*NPIX + n] = (_Float16)0.f;
}

// ======================= kernel 2: fused attention =========================
#define QT    128    // query rows per block (8 waves x 16 rows)
#define KT    32     // key columns per inner tile
#define NT    (NPIX / KT)
#define LDQ   136    // halves per K-row (256B data + 16B pad; 16B aligned)
#define LDV   40     // halves per V-row (64B data + 16B pad)
#define VROWS 144    // 132 real channels padded to 9*16 for WMMA ctiles
#define KBUF  (KT * LDQ)
#define VBUF  (VROWS * LDV)

// Build a 16-half WMMA fragment from a row base per ISA 7.12.2:
// lane<16 holds K = 0..7,16..23 (koff=0); lanes>=16 K = 8..15,24..31 (koff=8).
__device__ __forceinline__ v16h ldfrag(const _Float16* rowbase, int koff)
{
    v8h lo = *(const v8h*)(rowbase + koff);
    v8h hi = *(const v8h*)(rowbase + koff + 16);
    v16h f;
#pragma unroll
    for (int i = 0; i < 8; ++i) { f[i] = lo[i]; f[i+8] = hi[i]; }
    return f;
}

#if USE_TDM
typedef __attribute__((ext_vector_type(4))) unsigned int v4u;
typedef __attribute__((ext_vector_type(8))) int          v8i;
typedef __attribute__((ext_vector_type(4))) int          v4i;

// 2D TDM tile load, f16 elements, with LDS row padding (D# per ISA ch.8).
__device__ __forceinline__ void tdm_load_2d(
    unsigned lds_off, const _Float16* gptr,
    unsigned tensor_d0, unsigned tensor_d1, unsigned stride_e,
    unsigned tile_d0, unsigned tile_d1,
    unsigned pad_int_code, unsigned pad_amt_code)
{
    unsigned long long ga = (unsigned long long)gptr;
    v4u g0;
    g0[0] = 1u;                                     // count=1, user mode
    g0[1] = lds_off;                                // lds_addr
    g0[2] = (unsigned)(ga & 0xffffffffu);           // global_addr[31:0]
    g0[3] = (unsigned)((ga >> 32) & 0x01ffffffu)    // global_addr[56:32]
          | (2u << 30);                             // type = 2 (image)
    v8i g1;
    g1[0] = (int)((1u << 16)                        // data_size = 2B
                | (1u << 20)                        // pad_enable
                | (pad_int_code << 22)              // 2^(c+1) dwords per chunk
                | (pad_amt_code << 25));            // (c+1) dwords of pad
    g1[1] = (int)(tensor_d0 << 16);                 // tensor_dim0[15:0]
    g1[2] = (int)((tensor_d0 >> 16) | (tensor_d1 << 16));
    g1[3] = (int)((tensor_d1 >> 16) | (tile_d0 << 16));
    g1[4] = (int)(tile_d1 & 0xffffu);               // tile_dim1 (tile_dim2=0)
    g1[5] = (int)stride_e;                          // tensor_dim0_stride lo32
    g1[6] = 0;
    g1[7] = 0;
    v4i z4 = {0, 0, 0, 0};
#if __clang_major__ >= 23
    v8i z8 = {0, 0, 0, 0, 0, 0, 0, 0};
    __builtin_amdgcn_tensor_load_to_lds(g0, g1, z4, z4, z8, 0);
#else
    __builtin_amdgcn_tensor_load_to_lds(g0, g1, z4, z4, 0);
#endif
}
#endif

__global__ __launch_bounds__(256, 1) void attn_fused(
    const _Float16* __restrict__ Qh, const _Float16* __restrict__ Kh,
    const _Float16* __restrict__ Vt, float* __restrict__ out)
{
    __shared__ _Float16 sK[2 * KBUF];        // 17408 B (double-buffered)
    __shared__ _Float16 sV[2 * VBUF];        // 23040 B (double-buffered V^T)
    __shared__ _Float16 sP[8 * 16 * LDV];    // 10240 B (per-wave P staging)

    const int b     = blockIdx.y;
    const int q0    = blockIdx.x * QT;
    const int tid   = threadIdx.x;
    const int wv_   = tid >> 5;
    const int ln    = tid & 31;
    const int r     = ln & 15;
    const int koff  = (ln < 16) ? 0 : 8;
    const int rbase = (ln < 16) ? 0 : 8;

    // zero staging buffers once (visible LDS stores + zeroed V pad rows)
    {
        v8h z;
#pragma unroll
        for (int i = 0; i < 8; ++i) z[i] = (_Float16)0.f;
        for (int c = tid; c < 2*KBUF/8; c += 256) *(v8h*)&sK[c*8] = z;
        for (int c = tid; c < 2*VBUF/8; c += 256) *(v8h*)&sV[c*8] = z;
    }

    // Q fragments straight from global: 16 rows x K=128 (4 chunks of 32)
    const _Float16* qg = Qh + ((size_t)(b*NPIX + q0 + wv_*16 + r))*DCH;
    v16h qf[4];
#pragma unroll
    for (int kc = 0; kc < 4; ++kc) qf[kc] = ldfrag(qg + kc*32, koff);

    // accumulators: 9 ctiles (16x16 f32 C frags); pmax = local max of exp(s)
    v8f o[9];
#pragma unroll
    for (int ct = 0; ct < 9; ++ct)
#pragma unroll
        for (int i = 0; i < 8; ++i) o[ct][i] = 0.f;
    float pmax[8];
#pragma unroll
    for (int j = 0; j < 8; ++j) pmax[j] = 0.f;

    _Float16* sPw = &sP[wv_ * 16 * LDV];

#if USE_TDM
    __syncthreads();   // buffers zeroed before TDM engine writes them
    if (wv_ == 0) {    // prime tile 0 into buffer 0
        tdm_load_2d((unsigned)(unsigned long long)&sK[0],
                    Kh + (size_t)(b*NPIX)*DCH,
                    DCH, NPIX, DCH, DCH, KT, 5u, 3u);   // 64dw+4dw pad -> LDQ
        tdm_load_2d((unsigned)(unsigned long long)&sV[0],
                    Vt + (size_t)(b*VCH)*NPIX,
                    NPIX, VCH, NPIX, KT, VCH, 3u, 3u);  // 16dw+4dw pad -> LDV
    }
#endif

    for (int kt = 0; kt < NT; ++kt) {
        const int cur = kt & 1;
        const _Float16* cK = &sK[cur * KBUF];
        const _Float16* cV = &sV[cur * VBUF];

#if USE_TDM
        __builtin_amdgcn_s_wait_tensorcnt(0);   // no-op for waves != 0
        __syncthreads();                        // tile kt visible to all waves
        if (wv_ == 0 && kt + 1 < NT) {          // prefetch tile kt+1 via TDM
            const int nxt = (kt + 1) & 1, m1 = (kt + 1) * KT;
            tdm_load_2d((unsigned)(unsigned long long)&sK[nxt * KBUF],
                        Kh + ((size_t)(b*NPIX + m1))*DCH,
                        DCH, NPIX, DCH, DCH, KT, 5u, 3u);
            tdm_load_2d((unsigned)(unsigned long long)&sV[nxt * VBUF],
                        Vt + (size_t)(b*VCH)*NPIX + m1,
                        NPIX, VCH, NPIX, KT, VCH, 3u, 3u);
        }
#else
        const int m0 = kt * KT;
        __syncthreads();                        // previous tile consumed
        for (int c = tid; c < KT * (DCH/8); c += 256) {
            int row = c >> 4, c8 = (c & 15) * 8;
            *(v8h*)&sK[cur*KBUF + row*LDQ + c8] =
                *(const v8h*)(Kh + ((size_t)(b*NPIX + m0 + row))*DCH + c8);
        }
        for (int c = tid; c < VCH * (KT/8); c += 256) {
            int row = c >> 2, c8 = (c & 3) * 8;
            *(v8h*)&sV[cur*VBUF + row*LDV + c8] =
                *(const v8h*)(Vt + ((size_t)(b*VCH + row))*NPIX + m0 + c8);
        }
        if (kt + 1 < NT) {                      // global_prefetch_b8
            const char* nk = (const char*)(Kh + ((size_t)(b*NPIX + m0 + KT))*DCH);
            __builtin_prefetch(nk + tid*32, 0, 0);
            if (tid < VCH)
                __builtin_prefetch(Vt + ((size_t)(b*VCH + tid))*NPIX + m0 + KT, 0, 0);
        }
        __syncthreads();
#endif

        // ---- S = Q . K^T, two 16-wide key subtiles ----
        v8f sa, sb;
#pragma unroll
        for (int i = 0; i < 8; ++i) { sa[i] = 0.f; sb[i] = 0.f; }
#pragma unroll
        for (int kc = 0; kc < 4; ++kc) {
            v16h bfa = ldfrag(&cK[r        * LDQ] + kc*32, koff);
            v16h bfb = ldfrag(&cK[(16 + r) * LDQ] + kc*32, koff);
            sa = __builtin_amdgcn_wmma_f32_16x16x32_f16(false, qf[kc], false, bfa,
                                                        (short)0, sa, false, false);
            sb = __builtin_amdgcn_wmma_f32_16x16x32_f16(false, qf[kc], false, bfb,
                                                        (short)0, sb, false, false);
        }

        // ---- p = exp(s): bounded (|s|<=1), no max subtraction needed ----
#pragma unroll
        for (int j = 0; j < 8; ++j) {
            float pa = __expf(sa[j]);
            float pb = __expf(sb[j]);
            pmax[j] = fmaxf(pmax[j], fmaxf(pa, pb));
            sPw[(rbase + j)*LDV + r     ] = (_Float16)pa;    // C-frag -> LDS
            sPw[(rbase + j)*LDV + r + 16] = (_Float16)pb;
        }

        // ---- O += P . V_aug  (l accumulates via ones-column in ctile 8) ----
        v16h pf = ldfrag(&sPw[r * LDV], koff);
#pragma unroll
        for (int ct = 0; ct < 9; ++ct) {
            v16h bf = ldfrag(&cV[(ct*16 + r) * LDV], koff);
            o[ct] = __builtin_amdgcn_wmma_f32_16x16x32_f16(false, pf, false, bf,
                                                           (short)0, o[ct], false, false);
        }
    }

    // ---- epilogue: l from ones-column (col 2 of ctile 8); divide; max_score ----
#pragma unroll
    for (int j = 0; j < 8; ++j) {
        float pm = pmax[j];
#pragma unroll
        for (int msk = 1; msk < 16; msk <<= 1)     // one butterfly, end only
            pm = fmaxf(pm, __shfl_xor(pm, msk, 32));
        const float lj  = __shfl(o[8][j], (ln < 16) ? 2 : 18, 32);
        const float inv = 1.f / lj;
        const int   n   = q0 + wv_*16 + rbase + j;
#pragma unroll
        for (int ct = 0; ct < 8; ++ct) {
            const int cch = 128 + ct*16 + r;               // v1w channels
            out[((size_t)b*OUTC + cch)*NPIX + n] = o[ct][j] * inv;
        }
        if (r < 2)                                         // pos channels
            out[((size_t)b*OUTC + 256 + r)*NPIX + n] = o[8][j] * inv;
        if (r == 0)                                        // max_score = max(p)/l
            out[((size_t)b*OUTC + 258)*NPIX + n] = pm * inv;
    }
}

// ======================= launcher ==========================================
extern "C" void kernel_launch(void* const* d_in, const int* in_sizes, int n_in,
                              void* d_out, int out_size, void* d_ws, size_t ws_size,
                              hipStream_t stream)
{
    (void)in_sizes; (void)n_in; (void)out_size; (void)ws_size;

    const float* vol0 = (const float*)d_in[0];
    const float* vol1 = (const float*)d_in[1];
    const float* Wq   = (const float*)d_in[2];
    const float* Wk   = (const float*)d_in[3];
    const float* Wv   = (const float*)d_in[4];
    float*       out  = (float*)d_out;

    // workspace layout (f16): Q | K | V^T  (~25.4 MB total)
    _Float16* Qh = (_Float16*)d_ws;
    _Float16* Kh = Qh + (size_t)BATCH * NPIX * DCH;
    _Float16* Vt = Kh + (size_t)BATCH * NPIX * DCH;

    dim3 gp(NPIX / PREP_TILE, BATCH);
    qkv_prep<<<gp, dim3(PREP_TILE), 0, stream>>>(vol0, vol1, Wq, Wk, Wv,
                                                 Qh, Kh, Vt, out);

    dim3 ga(NPIX / QT, BATCH);
    attn_fused<<<ga, dim3(256), 0, stream>>>(Qh, Kh, Vt, out);
}
